// GatedDeltaNetAttention_32839319945662
// MI455X (gfx1250) — matrix operand order repaired and checked by reading, not compiled, not run on hardware
//
#include <hip/hip_runtime.h>
#include <math.h>

// ---------------------------------------------------------------------------
// Gated DeltaNet forward for MI455X (gfx1250), wave32.
// GEMMs: v_wmma_f32_16x16x32_bf16 fed by Tensor-Data-Mover DMA into LDS
// (double buffered, TENSORcnt synchronized). Scan is fp32 VALU.
// ---------------------------------------------------------------------------

typedef __bf16 v16bf __attribute__((ext_vector_type(16)));
typedef __bf16 v8bf  __attribute__((ext_vector_type(8)));
typedef float  v8f   __attribute__((ext_vector_type(8)));
typedef unsigned int v4u __attribute__((ext_vector_type(4)));
typedef int v8i __attribute__((ext_vector_type(8)));
typedef int v4i __attribute__((ext_vector_type(4)));

#define BB 4
#define SS 2048
#define HH 1024
#define NHH 16
#define HDD 64
#define MM (BB*SS)            // 8192 rows

#if defined(__has_builtin)
#if __has_builtin(__builtin_amdgcn_tensor_load_to_lds) && \
    __has_builtin(__builtin_amdgcn_s_wait_tensorcnt)
#define USE_TDM 1
#endif
#endif
#ifndef USE_TDM
#define USE_TDM 0
#endif

// ---------------- fp32 -> bf16 (round to nearest even) ----------------
__global__ void cast_f32_bf16_kernel(const float* __restrict__ src,
                                     unsigned short* __restrict__ dst, int n) {
  int i = blockIdx.x * blockDim.x + threadIdx.x;
  if (i < n) {
    unsigned int u = __float_as_uint(src[i]);
    unsigned int r = u + 0x7fffu + ((u >> 16) & 1u);
    dst[i] = (unsigned short)(r >> 16);
  }
}

#if USE_TDM
// 2D tile DMA: global (row-major, row_stride elems, 2-byte elems) -> LDS.
// D# packing per CDNA5 ISA (groups 2/3 zero = 2D).
// This toolchain exposes the 6-arg builtin: (g0, g1, g2, g3, g_extra, cpol).
__device__ __forceinline__ void tdm_load_tile_2d(const void* gsrc,
                                                 unsigned lds_byte_off,
                                                 unsigned tile_k,      // dim0 elems
                                                 unsigned tile_rows,   // dim1 rows
                                                 unsigned tensor_k,    // row length
                                                 unsigned row_stride)  // elems
{
  unsigned long long ga = (unsigned long long)gsrc;
  v4u g0;
  g0[0] = 1u;                                                // count=1
  g0[1] = lds_byte_off;                                      // lds_addr
  g0[2] = (unsigned)ga;                                      // global_addr[31:0]
  g0[3] = (unsigned)((ga >> 32) & 0x01ffffffu) | (2u << 30); // addr[56:32]|type=2
  v8i g1;
  g1[0] = (int)(1u << 16);                                   // data_size=2B
  g1[1] = (int)((tensor_k & 0xffffu) << 16);                 // tensor_dim0 lo
  g1[2] = (int)((tensor_k >> 16) | ((tile_rows & 0xffffu) << 16)); // dim0 hi | dim1 lo
  g1[3] = (int)(((tile_rows >> 16) & 0xffffu) | (tile_k << 16));   // dim1 hi | tile_dim0
  g1[4] = (int)(tile_rows & 0xffffu);                        // tile_dim1 (tile_dim2=0)
  g1[5] = (int)row_stride;                                   // tensor_dim0_stride lo
  g1[6] = 0;                                                 // stride hi | dim1_stride lo
  g1[7] = 0;
  v4i z4 = {0, 0, 0, 0};
  v8i z8 = {0, 0, 0, 0, 0, 0, 0, 0};
  __builtin_amdgcn_tensor_load_to_lds(g0, g1, z4, z4, z8, 0);
}
#endif

// ---------------- WMMA GEMM: Y[M,N] = X[M,K] * W[N,K]^T  (bf16->f32) ----
// block = (32,8): 8 waves. Block tile 64(M) x 128(N), K-step 32.
// Wave w: rows (w>>2)*32..+31, cols (w&3)*32..+31 -> 2x2 wmma tiles.
// A/B tiles staged in LDS (double buffered); TDM DMA when available.
__global__ __launch_bounds__(256)
void gemm_bf16_wmma_kernel(const __bf16* __restrict__ X,
                           const __bf16* __restrict__ W,
                           float* __restrict__ Y,
                           int M, int N, int K) {
  __shared__ __attribute__((aligned(16))) __bf16 As[2][64][32];   // 2 x 4KB
  __shared__ __attribute__((aligned(16))) __bf16 Bs[2][128][32];  // 2 x 8KB

  const int lane = threadIdx.x;          // 0..31
  const int wave = threadIdx.y;          // 0..7
  const int r15  = lane & 15;
  const int hi   = lane >> 4;            // 0|1
  const int wm   = wave >> 2;            // 0..1  (M sub-block of 32)
  const int wn   = wave & 3;             // 0..3  (N sub-block of 32)
  const int tid  = wave * 32 + lane;
  const int blockM = blockIdx.y * 64;
  const int blockN = blockIdx.x * 128;

  const __bf16* Ag = X + (size_t)blockM * K;   // A tile origin (rows blockM..+63)
  const __bf16* Bg = W + (size_t)blockN * K;   // B tile origin (W rows blockN..+127)

#if USE_TDM
  const bool leader = (wave == 0);
  const unsigned aLds0 = (unsigned)(unsigned long long)(const void*)&As[0][0][0];
  const unsigned aLds1 = (unsigned)(unsigned long long)(const void*)&As[1][0][0];
  const unsigned bLds0 = (unsigned)(unsigned long long)(const void*)&Bs[0][0][0];
  const unsigned bLds1 = (unsigned)(unsigned long long)(const void*)&Bs[1][0][0];
#endif

  // ---- stage K-chunk k0 into buffer p ----
#if USE_TDM
#define STAGE(p, k0)                                                          \
  do {                                                                        \
    if (leader) {                                                             \
      tdm_load_tile_2d(Ag + (k0), (p) ? aLds1 : aLds0, 32u, 64u,  (unsigned)K, (unsigned)K); \
      tdm_load_tile_2d(Bg + (k0), (p) ? bLds1 : bLds0, 32u, 128u, (unsigned)K, (unsigned)K); \
    }                                                                         \
  } while (0)
#define STAGE_FINISH()                                                        \
  do { if (leader) __builtin_amdgcn_s_wait_tensorcnt(0); } while (0)
#else
#define STAGE(p, k0)                                                          \
  do {                                                                        \
    /* A: 256 16-byte chunks; B: 512 chunks */                                \
    {                                                                         \
      int c = tid, row = c >> 2, c8 = (c & 3) * 8;                            \
      *(v8bf*)&As[p][row][c8] = *(const v8bf*)(Ag + (size_t)row * K + (k0) + c8); \
    }                                                                         \
    for (int c = tid; c < 512; c += 256) {                                    \
      int row = c >> 2, c8 = (c & 3) * 8;                                     \
      *(v8bf*)&Bs[p][row][c8] = *(const v8bf*)(Bg + (size_t)row * K + (k0) + c8); \
    }                                                                         \
  } while (0)
#define STAGE_FINISH() do { } while (0)
#endif

  v8f acc00 = {}, acc01 = {}, acc10 = {}, acc11 = {};

  STAGE(0, 0);
  STAGE_FINISH();
  __syncthreads();

  int cur = 0;
  for (int k0 = 0; k0 < K; k0 += 32) {
    const int nxt = cur ^ 1;
    const bool more = (k0 + 32) < K;
    if (more) STAGE(nxt, k0 + 32);

    // A fragments (16x32): lane row = mt*16+r15; halves 0..7 K=hi*8+j, 8..15 K=16+hi*8+j
    v16bf Af[2];
#pragma unroll
    for (int mi = 0; mi < 2; ++mi) {
      int row = (wm * 2 + mi) * 16 + r15;
      v8bf a0 = *(const v8bf*)&As[cur][row][hi * 8];
      v8bf a1 = *(const v8bf*)&As[cur][row][16 + hi * 8];
#pragma unroll
      for (int j = 0; j < 8; ++j) { Af[mi][j] = a0[j]; Af[mi][j + 8] = a1[j]; }
    }
    // B fragments (32x16): lane col = nt*16+r15; halves 0..7 K=hi*8+j, 8..15 K=16+hi*8+j
    v16bf Bf[2];
#pragma unroll
    for (int ni = 0; ni < 2; ++ni) {
      int row = (wn * 2 + ni) * 16 + r15;
      v8bf b0 = *(const v8bf*)&Bs[cur][row][hi * 8];
      v8bf b1 = *(const v8bf*)&Bs[cur][row][16 + hi * 8];
#pragma unroll
      for (int j = 0; j < 8; ++j) { Bf[ni][j] = b0[j]; Bf[ni][j + 8] = b1[j]; }
    }

    acc00 = __builtin_amdgcn_wmma_f32_16x16x32_bf16(false, Af[0], false, Bf[0], (short)0, acc00, false, false);
    acc01 = __builtin_amdgcn_wmma_f32_16x16x32_bf16(false, Af[0], false, Bf[1], (short)0, acc01, false, false);
    acc10 = __builtin_amdgcn_wmma_f32_16x16x32_bf16(false, Af[1], false, Bf[0], (short)0, acc10, false, false);
    acc11 = __builtin_amdgcn_wmma_f32_16x16x32_bf16(false, Af[1], false, Bf[1], (short)0, acc11, false, false);

    if (more) STAGE_FINISH();
    __syncthreads();
    cur = nxt;
  }

  // D layout: VGPR r -> row = tileM*16 + hi*8 + r ; col = tileN*16 + r15
  v8f accs[2][2] = {{acc00, acc01}, {acc10, acc11}};
#pragma unroll
  for (int mi = 0; mi < 2; ++mi) {
#pragma unroll
    for (int ni = 0; ni < 2; ++ni) {
      int rowb = blockM + (wm * 2 + mi) * 16 + hi * 8;
      int col  = blockN + (wn * 2 + ni) * 16 + r15;
      float* yBase = Y + (size_t)rowb * N + col;
#pragma unroll
      for (int r = 0; r < 8; ++r) yBase[(size_t)r * N] = accs[mi][ni][r];
    }
  }
#undef STAGE
#undef STAGE_FINISH
}

// ---------------- alpha/beta = sigmoid(x @ W{a,b}^T + b{a,b}) ----------
__global__ __launch_bounds__(256)
void alphabeta_kernel(const float* __restrict__ x,
                      const float* __restrict__ Wa, const float* __restrict__ ba,
                      const float* __restrict__ Wb, const float* __restrict__ bb,
                      float* __restrict__ alpha, float* __restrict__ beta) {
  int idx = blockIdx.x * blockDim.x + threadIdx.x;   // row*16 + h
  if (idx >= MM * NHH) return;
  int row = idx >> 4, h = idx & 15;
  const float* xr = x  + (size_t)row * HH;
  const float* wa = Wa + (size_t)h * HH;
  const float* wb = Wb + (size_t)h * HH;
  float sa = 0.f, sb = 0.f;
  for (int k = 0; k < HH; ++k) {
    float xv = xr[k];
    sa = fmaf(xv, wa[k], sa);
    sb = fmaf(xv, wb[k], sb);
  }
  sa += ba[h]; sb += bb[h];
  alpha[idx] = 1.f / (1.f + __expf(-sa));
  beta[idx]  = 1.f / (1.f + __expf(-sb));
}

// ---------------- per-head l2norm on q,k ; silu on v -------------------
__global__ __launch_bounds__(64)
void head_act_kernel(float* __restrict__ q, float* __restrict__ k,
                     float* __restrict__ v) {
  __shared__ float red[64];
  int hidx = blockIdx.x;                 // (b*S+s)*NH + h
  int d = threadIdx.x;
  size_t base = (size_t)hidx * HDD + d;
  float qv = q[base], kv = k[base], vv = v[base];

  red[d] = qv * qv; __syncthreads();
  for (int off = 32; off > 0; off >>= 1) {
    if (d < off) red[d] += red[d + off];
    __syncthreads();
  }
  float qn = sqrtf(red[0]); __syncthreads();

  red[d] = kv * kv; __syncthreads();
  for (int off = 32; off > 0; off >>= 1) {
    if (d < off) red[d] += red[d + off];
    __syncthreads();
  }
  float kn = sqrtf(red[0]);

  q[base] = qv / fmaxf(qn, 1e-12f);
  k[base] = kv / fmaxf(kn, 1e-12f);
  v[base] = vv / (1.f + __expf(-vv));    // silu
}

// ---------------- recurrent delta-rule scan ----------------------------
// one block (256 threads) per (b, head). state[64][64] in registers:
// thread owns row d = tid>>2, 16 columns e = (tid&3)*16 .. +15.
__global__ __launch_bounds__(256)
void scan_kernel(const float* __restrict__ q, const float* __restrict__ k,
                 const float* __restrict__ v,
                 const float* __restrict__ alpha, const float* __restrict__ beta,
                 float* __restrict__ out) {
  const int bh  = blockIdx.x;            // b*NH + h
  const int b   = bh >> 4, h = bh & 15;
  const int tid = threadIdx.x;
  const int d   = tid >> 2;
  const int eg  = (tid & 3) * 16;
  __shared__ float ks[64], vs[64], qs[64], ab[2];

  float s[16];
#pragma unroll
  for (int j = 0; j < 16; ++j) s[j] = 0.f;

  const size_t base = (((size_t)b * SS) * NHH + h) * HDD;  // token stride = HH
  for (int t = 0; t < SS; ++t) {
    size_t off = base + (size_t)t * HH;
    if (tid < 64)        ks[tid]       = k[off + tid];
    else if (tid < 128)  vs[tid - 64]  = v[off + tid - 64];
    else if (tid < 192)  qs[tid - 128] = q[off + tid - 128];
    else if (tid == 192) {
      size_t abo = ((size_t)b * SS + t) * NHH + h;
      ab[0] = alpha[abo]; ab[1] = beta[abo];
    }
    __syncthreads();
    const float a = ab[0], be = ab[1];
    const float kd = ks[d], vd = vs[d];
    float p = 0.f;
#pragma unroll
    for (int j = 0; j < 16; ++j) {
      float ke  = ks[eg + j];
      float sij = s[j];
      sij  = a * sij * (1.f - be * kd * ke) + be * vd * ke;
      s[j] = sij;
      p = fmaf(sij, qs[eg + j], p);
    }
    p += __shfl_xor(p, 1, 32);
    p += __shfl_xor(p, 2, 32);
    if ((tid & 3) == 0) out[off + d] = p;
    __syncthreads();
  }
}

// ---------------- LayerNorm + silu gate, emit bf16 ---------------------
__global__ __launch_bounds__(256)
void ln_gate_kernel(const float* __restrict__ attn,
                    const float* __restrict__ gatepre,
                    const float* __restrict__ ln_g, const float* __restrict__ ln_b,
                    unsigned short* __restrict__ gated) {
  __shared__ float red[256];
  int row = blockIdx.x, tid = threadIdx.x;
  const float* ar = attn + (size_t)row * HH;
  float vals[4], sum = 0.f, sumsq = 0.f;
#pragma unroll
  for (int j = 0; j < 4; ++j) {
    float x = ar[tid + j * 256];
    vals[j] = x; sum += x; sumsq = fmaf(x, x, sumsq);
  }
  red[tid] = sum; __syncthreads();
  for (int off = 128; off > 0; off >>= 1) {
    if (tid < off) red[tid] += red[tid + off];
    __syncthreads();
  }
  float mu = red[0] * (1.f / HH); __syncthreads();
  red[tid] = sumsq; __syncthreads();
  for (int off = 128; off > 0; off >>= 1) {
    if (tid < off) red[tid] += red[tid + off];
    __syncthreads();
  }
  float var  = red[0] * (1.f / HH) - mu * mu;
  float rstd = rsqrtf(var + 1e-5f);
#pragma unroll
  for (int j = 0; j < 4; ++j) {
    int c = tid + j * 256;
    float y = (vals[j] - mu) * rstd * ln_g[c] + ln_b[c];
    float g = gatepre[(size_t)row * HH + c];
    y *= g / (1.f + __expf(-g));                  // * silu(gate)
    unsigned int u = __float_as_uint(y);
    unsigned int r = u + 0x7fffu + ((u >> 16) & 1u);
    gated[(size_t)row * HH + c] = (unsigned short)(r >> 16);
  }
}

// ---------------------------------------------------------------------------
extern "C" void kernel_launch(void* const* d_in, const int* in_sizes, int n_in,
                              void* d_out, int out_size, void* d_ws, size_t ws_size,
                              hipStream_t stream) {
  (void)in_sizes; (void)n_in; (void)out_size; (void)ws_size;
  const float* x    = (const float*)d_in[0];
  const float* Wq   = (const float*)d_in[1];
  const float* Wk   = (const float*)d_in[2];
  const float* Wv   = (const float*)d_in[3];
  const float* Wa   = (const float*)d_in[4];
  const float* ba   = (const float*)d_in[5];
  const float* Wb   = (const float*)d_in[6];
  const float* bb   = (const float*)d_in[7];
  const float* Wg   = (const float*)d_in[8];
  const float* Wo   = (const float*)d_in[9];
  const float* ln_g = (const float*)d_in[10];
  const float* ln_b = (const float*)d_in[11];
  float* out = (float*)d_out;

  // -------- workspace layout --------
  char* ws = (char*)d_ws;
  size_t off = 0;
  auto carve = [&](size_t bytes) { char* p = ws + off; off += (bytes + 255) & ~(size_t)255; return p; };
  const size_t nX  = (size_t)MM * HH;
  const size_t nW  = (size_t)HH * HH;
  unsigned short* xb  = (unsigned short*)carve(nX * 2);
  unsigned short* wqb = (unsigned short*)carve(nW * 2);
  unsigned short* wkb = (unsigned short*)carve(nW * 2);
  unsigned short* wvb = (unsigned short*)carve(nW * 2);
  unsigned short* wgb = (unsigned short*)carve(nW * 2);
  unsigned short* wob = (unsigned short*)carve(nW * 2);
  float* qf  = (float*)carve(nX * 4);
  float* kf  = (float*)carve(nX * 4);
  float* vf  = (float*)carve(nX * 4);
  float* gp  = (float*)carve(nX * 4);
  float* ao  = (float*)carve(nX * 4);
  float* al  = (float*)carve((size_t)MM * NHH * 4);
  float* be  = (float*)carve((size_t)MM * NHH * 4);
  unsigned short* gated = (unsigned short*)carve(nX * 2);

  // -------- 1. bf16 conversions --------
  cast_f32_bf16_kernel<<<(int)((nX + 255) / 256), 256, 0, stream>>>(x,  xb,  (int)nX);
  cast_f32_bf16_kernel<<<(int)((nW + 255) / 256), 256, 0, stream>>>(Wq, wqb, (int)nW);
  cast_f32_bf16_kernel<<<(int)((nW + 255) / 256), 256, 0, stream>>>(Wk, wkb, (int)nW);
  cast_f32_bf16_kernel<<<(int)((nW + 255) / 256), 256, 0, stream>>>(Wv, wvb, (int)nW);
  cast_f32_bf16_kernel<<<(int)((nW + 255) / 256), 256, 0, stream>>>(Wg, wgb, (int)nW);
  cast_f32_bf16_kernel<<<(int)((nW + 255) / 256), 256, 0, stream>>>(Wo, wob, (int)nW);

  // -------- 2. big WMMA GEMMs (TDM->LDS staged) --------
  dim3 gblk(32, 8);
  dim3 ggrd(HH / 128, MM / 64);              // (8, 128)
  gemm_bf16_wmma_kernel<<<ggrd, gblk, 0, stream>>>((const __bf16*)xb, (const __bf16*)wqb, qf, MM, HH, HH);
  gemm_bf16_wmma_kernel<<<ggrd, gblk, 0, stream>>>((const __bf16*)xb, (const __bf16*)wkb, kf, MM, HH, HH);
  gemm_bf16_wmma_kernel<<<ggrd, gblk, 0, stream>>>((const __bf16*)xb, (const __bf16*)wvb, vf, MM, HH, HH);
  gemm_bf16_wmma_kernel<<<ggrd, gblk, 0, stream>>>((const __bf16*)xb, (const __bf16*)wgb, gp, MM, HH, HH);

  // -------- 3. alpha/beta --------
  alphabeta_kernel<<<(MM * NHH + 255) / 256, 256, 0, stream>>>(x, Wa, ba, Wb, bb, al, be);

  // -------- 4. l2norm(q), l2norm(k), silu(v) --------
  head_act_kernel<<<MM * NHH, 64, 0, stream>>>(qf, kf, vf);

  // -------- 5. recurrent delta-rule scan --------
  scan_kernel<<<BB * NHH, 256, 0, stream>>>(qf, kf, vf, al, be, ao);

  // -------- 6. LayerNorm + silu gate -> bf16 --------
  ln_gate_kernel<<<MM, 256, 0, stream>>>(ao, gp, ln_g, ln_b, gated);

  // -------- 7. final WMMA GEMM --------
  gemm_bf16_wmma_kernel<<<ggrd, gblk, 0, stream>>>((const __bf16*)gated, (const __bf16*)wob, out, MM, HH, HH);
}
